// KDE_67370857005236
// MI455X (gfx1250) — compile-verified, matrix operand order checked by reading, not run on hardware
//
#include <hip/hip_runtime.h>
#include <math.h>

typedef __attribute__((ext_vector_type(16))) _Float16 v16h;
typedef __attribute__((ext_vector_type(8)))  float    v8f;

#define LOG_2PI_F 1.8378770664093453f
#define LOG2E_F   1.4426950408889634f
#define TILE 16

#if __has_builtin(__builtin_amdgcn_exp2f)
#define EXP2F(x) __builtin_amdgcn_exp2f(x)
#else
#define EXP2F(x) exp2f(x)
#endif

__global__ void kde_zero_out(float* __restrict__ out, int n) {
    int i = blockIdx.x * blockDim.x + threadIdx.x;
    if (i < n) out[i] = 0.0f;
}

// ---------------------------------------------------------------------------
// Pack B fragments once: for base point n, the v16h fragment a lane needs is
//   half 0: [ c*-2*xb[0..7],  c, c*sqb, 0 x 6 ]   (K = 0..7, 16, 17)
//   half 1: [ c*-2*xb[8..15], 0 x 8 ]             (K = 8..15, 24..31)
// c = -log2(e)/(2*bw^2) folded in, so WMMA output is directly the exp2 arg.
// Layout: bpack[n*64B + half*32B]; 1 MB total, L2-resident.
// ---------------------------------------------------------------------------
__global__ void kde_pack_b(const float* __restrict__ xb_g,
                           const float* __restrict__ log_bw_p,
                           _Float16* __restrict__ bpack, int N) {
    int n = blockIdx.x * blockDim.x + threadIdx.x;
    if (n >= N) return;
    const float log_bw = log_bw_p[0];
    const float inv = 0.5f * __expf(-2.0f * log_bw);   // 1/(2*bw^2)
    const float c   = -inv * LOG2E_F;

    const float* row = xb_g + (size_t)n * 16;
    float x[16];
    float sq = 0.0f;
#pragma unroll
    for (int k = 0; k < 16; ++k) { x[k] = row[k]; sq = fmaf(x[k], x[k], sq); }

    _Float16* o = bpack + (size_t)n * 32;
#pragma unroll
    for (int e = 0; e < 8; ++e) o[e] = (_Float16)(c * -2.0f * x[e]);
    o[8] = (_Float16)c;                 // multiplies A's sqe slot (K=16)
    o[9] = (_Float16)(c * sq);          // multiplies A's 1.0 slot (K=17)
#pragma unroll
    for (int e = 10; e < 16; ++e) o[e] = (_Float16)0.0f;
#pragma unroll
    for (int e = 0; e < 8; ++e) o[16 + e] = (_Float16)(c * -2.0f * x[8 + e]);
#pragma unroll
    for (int e = 24; e < 32; ++e) o[e] = (_Float16)0.0f;
}

// Shared helpers -------------------------------------------------------------
__device__ __forceinline__ v16h build_a_frag(const float* __restrict__ xe_g,
                                             int m, int half) {
    const float* rowE = xe_g + (size_t)m * 16;
    float xe[16];
    float sqe = 0.0f;
#pragma unroll
    for (int k = 0; k < 16; ++k) { xe[k] = rowE[k]; sqe = fmaf(xe[k], xe[k], sqe); }
    v16h a;
#pragma unroll
    for (int e = 0; e < 8; ++e) a[e] = (_Float16)xe[8 * half + e];   // K = 8h..8h+7
#pragma unroll
    for (int e = 8; e < 16; ++e) a[e] = (_Float16)0.0f;
    if (half == 0) { a[8] = (_Float16)sqe; a[9] = (_Float16)1.0f; }  // K=16,17
    return a;
}

__device__ __forceinline__ void reduce_and_store(float (&acc)[8], float coef,
                                                 float* __restrict__ out,
                                                 int m0, int half, int lr) {
#pragma unroll
    for (int v = 0; v < 8; ++v) {
        float s = acc[v];
#pragma unroll
        for (int off = 1; off < 16; off <<= 1)
            s += __shfl_xor(s, off, 32);
        if (lr == 0)
            atomicAdd(&out[m0 + half * 8 + v], coef * s);
    }
}

// ---------------------------------------------------------------------------
// Fast path: double-buffered software pipeline, scalar loop control.
// Steady state per 2 tiles: 2 prefetch + 4 b128 loads + 2 WMMA + 16 exp + 8 pk_add.
// ---------------------------------------------------------------------------
__global__ void __launch_bounds__(256)
kde_wmma_packed(const float* __restrict__ xe_g,
                const _Float16* __restrict__ bpack,
                const float* __restrict__ log_bw_p,
                float* __restrict__ out,
                float invN, int N, int nchunks) {
    const int lane = threadIdx.x & 31;
    const int half = lane >> 4;
    const int lr   = lane & 15;
    const int m0   = blockIdx.x * TILE;
    // wave-uniform chunk id -> SGPR so loop control is scalar
    const int chunk = __builtin_amdgcn_readfirstlane(
        blockIdx.y * blockDim.y + threadIdx.y);

    const float log_bw = log_bw_p[0];
    const float lt1    = -0.5f * 16.0f * LOG_2PI_F - log_bw;
    const float coef   = __expf(lt1) * invN;

    const v16h a = build_a_frag(xe_g, m0 + lr, half);

    float acc[8];
#pragma unroll
    for (int v = 0; v < 8; ++v) acc[v] = 0.0f;

    const int ntiles = N / TILE;
    const int tpc    = (ntiles + nchunks - 1) / nchunks;
    const int tbeg   = chunk * tpc;
    int tend = tbeg + tpc; if (tend > ntiles) tend = ntiles;
    const int cnt = tend - tbeg;

    auto consume = [&](v16h bb) {
        v8f z = {0.f, 0.f, 0.f, 0.f, 0.f, 0.f, 0.f, 0.f};
        v8f e2 = __builtin_amdgcn_wmma_f32_16x16x32_f16(
            false, a, false, bb, (short)0, z, false, false);  // e2 = c*d2
#pragma unroll
        for (int v = 0; v < 8; ++v)
            acc[v] += EXP2F(e2[v]);                           // v_exp_f32 direct
    };

    if (cnt > 0) {
        // lane's fragment: base + n*64B + half*32B; tiles are 1024B apart
        const char* p = (const char*)bpack
                      + (size_t)(tbeg * TILE + lr) * 64 + (size_t)half * 32;
        v16h b0 = *(const v16h*)p;          // tile i
        v16h b1{};

        int i = 0;
        for (; i + 2 <= cnt - 1; i += 2) {  // consume tiles i, i+1; preload i+1, i+2
            __builtin_prefetch(p + 8192, 0, 3);
            __builtin_prefetch(p + 9216, 0, 3);
            b1 = *(const v16h*)(p + 1024);  // imm-offset load
            consume(b0);
            b0 = *(const v16h*)(p + 2048);  // imm-offset load
            consume(b1);
            p += 2048;
        }
        if (i == cnt - 2) {                 // two tiles left
            b1 = *(const v16h*)(p + 1024);
            consume(b0);
            consume(b1);
        } else {                            // one tile left
            consume(b0);
        }
    }

    reduce_and_store(acc, coef, out, m0, half, lr);
}

// ---------------------------------------------------------------------------
// Fallback (ws too small): build B fragments on the fly.
// ---------------------------------------------------------------------------
__global__ void __launch_bounds__(256)
kde_wmma_fly(const float* __restrict__ xe_g,
             const float* __restrict__ xb_g,
             const float* __restrict__ log_bw_p,
             float* __restrict__ out,
             float invN, int N, int nchunks) {
    const int lane = threadIdx.x & 31;
    const int half = lane >> 4;
    const int lr   = lane & 15;
    const int m0   = blockIdx.x * TILE;
    const int chunk = __builtin_amdgcn_readfirstlane(
        blockIdx.y * blockDim.y + threadIdx.y);

    const float log_bw = log_bw_p[0];
    const float inv  = 0.5f * __expf(-2.0f * log_bw);
    const float c    = -inv * LOG2E_F;
    const float lt1  = -0.5f * 16.0f * LOG_2PI_F - log_bw;
    const float coef = __expf(lt1) * invN;

    const v16h a = build_a_frag(xe_g, m0 + lr, half);

    float acc[8];
#pragma unroll
    for (int v = 0; v < 8; ++v) acc[v] = 0.0f;

    const int ntiles = N / TILE;
    const int tpc    = (ntiles + nchunks - 1) / nchunks;
    const int tbeg   = chunk * tpc;
    int tend = tbeg + tpc; if (tend > ntiles) tend = ntiles;

    for (int t = tbeg; t < tend; ++t) {
        const int n = t * TILE + lr;
        const float* rowB = xb_g + (size_t)n * 16;
        float xb[16];
        float sqb = 0.0f;
#pragma unroll
        for (int k = 0; k < 16; ++k) { xb[k] = rowB[k]; sqb = fmaf(xb[k], xb[k], sqb); }
        v16h b;
#pragma unroll
        for (int e = 0; e < 8; ++e) b[e] = (_Float16)(c * -2.0f * xb[8 * half + e]);
#pragma unroll
        for (int e = 8; e < 16; ++e) b[e] = (_Float16)0.0f;
        if (half == 0) { b[8] = (_Float16)c; b[9] = (_Float16)(c * sqb); }

        v8f z = {0.f, 0.f, 0.f, 0.f, 0.f, 0.f, 0.f, 0.f};
        v8f e2 = __builtin_amdgcn_wmma_f32_16x16x32_f16(
            false, a, false, b, (short)0, z, false, false);
#pragma unroll
        for (int v = 0; v < 8; ++v)
            acc[v] += EXP2F(e2[v]);
    }

    reduce_and_store(acc, coef, out, m0, half, lr);
}

extern "C" void kernel_launch(void* const* d_in, const int* in_sizes, int n_in,
                              void* d_out, int out_size, void* d_ws, size_t ws_size,
                              hipStream_t stream) {
    const float* x_eval = (const float*)d_in[0];
    const float* x_base = (const float*)d_in[1];
    const float* log_bw = (const float*)d_in[2];
    float* out = (float*)d_out;

    const int M = in_sizes[0] / 16;   // 8192
    const int N = in_sizes[1] / 16;   // 16384
    const float invN = 1.0f / (float)N;

    kde_zero_out<<<(out_size + 255) / 256, 256, 0, stream>>>(out, out_size);

    const int WAVES_PER_BLOCK = 8;    // 256 threads = 8 wave32
    const int CHUNK_BLOCKS    = 4;    // nchunks = 32 -> 16384 waves total
    dim3 block(32, WAVES_PER_BLOCK);
    dim3 grid(M / TILE, CHUNK_BLOCKS);
    const int nchunks = CHUNK_BLOCKS * WAVES_PER_BLOCK;

    const size_t need = (size_t)N * 32 * 2 + 16384;  // packed B + prefetch slack
    if (ws_size >= need) {
        _Float16* bpack = (_Float16*)d_ws;
        kde_pack_b<<<(N + 255) / 256, 256, 0, stream>>>(x_base, log_bw, bpack, N);
        kde_wmma_packed<<<grid, block, 0, stream>>>(x_eval, bpack, log_bw, out,
                                                    invN, N, nchunks);
    } else {
        kde_wmma_fly<<<grid, block, 0, stream>>>(x_eval, x_base, log_bw, out,
                                                 invN, N, nchunks);
    }
}
